// sLSTM_63496796504571
// MI455X (gfx1250) — matrix operand order repaired and checked by reading, not compiled, not run on hardware
//
#include <hip/hip_runtime.h>
#include <hip/hip_bf16.h>

// ---------------------------------------------------------------------------
// Persistent 2-layer sLSTM for MI455X (gfx1250, wave32, WMMA bf16).
//  * 32 workgroups x 256 threads (8 waves). WG w owns hidden cols [16w,16w+16).
//  * All weight slices converted to bf16 and cached in LDS (285KB <= 320KB),
//    XOR-swizzled in 16B chunks for conflict-free ds_read_b128 B-fragments.
//  * Gate GEMMs via v_wmma_f32_16x16x32_bf16; c-state lives in VGPRs.
//  * h published per step to L2-resident double-buffered bf16 arrays;
//    2 grid barriers per timestep (atomic generation barrier).
//  * Concatenated-K GEMMs split into homogeneous sub-loops (no inner branch).
// ---------------------------------------------------------------------------

typedef __attribute__((ext_vector_type(16))) __bf16        v16bf;
typedef __attribute__((ext_vector_type(8)))  float         v8f;
typedef __attribute__((ext_vector_type(4)))  unsigned int  u32x4;
typedef unsigned short ushort_t;

static constexpr int T_STEPS = 1024;
static constexpr int BATCH   = 64;
static constexpr int IDIM    = 256;
static constexpr int HDIM    = 512;
static constexpr int NWG     = 32;     // one WG per WGP, all resident
static constexpr int HS      = 16;     // hidden cols per WG
static constexpr int NTHR    = 256;    // 8 wave32

// LDS chunk counts (1 chunk = 16B = 8 bf16)
static constexpr int W0_CH = (IDIM + HDIM) * 2 / 16;   // 96  ([Wih0|Whh0] row)
static constexpr int W1_CH = (HDIM + HDIM) * 2 / 16;   // 128 ([Wih1|Whh1] row)
static constexpr int WF_CH = HDIM * 2 / 16;            // 64

static constexpr int SW0_US = 64 * W0_CH * 8;          // ushort counts
static constexpr int SW1_US = 64 * W1_CH * 8;
static constexpr int SWF_US = 16 * WF_CH * 8;
static constexpr size_t LDS_BYTES =
    (size_t)(SW0_US + SW1_US + 2 * SWF_US) * 2 +       // 262144 B weights
    (4 * 64 * 17 + 64 * 17) * 4;                       // gate + f staging

// fp32 -> bf16 round-to-nearest-even (bit trick)
__device__ __forceinline__ ushort_t f2bf(float f) {
  union { float f; unsigned u; } v; v.f = f;
  unsigned r = v.u + 0x7FFFu + ((v.u >> 16) & 1u);
  return (ushort_t)(r >> 16);
}
__device__ __forceinline__ float sigf(float x) { return 1.f / (1.f + __expf(-x)); }

// Store one 16B bf16 chunk into a swizzled LDS weight row.
__device__ __forceinline__ void store_wchunk(ushort_t* rowp, int c, int row,
                                             const float* __restrict__ src) {
  union { ushort_t s[8]; u32x4 q; } tmp;
#pragma unroll
  for (int j = 0; j < 8; ++j) tmp.s[j] = f2bf(src[j]);
  *(u32x4*)(rowp + ((c ^ (row & 15)) << 3)) = tmp.q;
}

// B fragment (32x16 bf16): lane n=lane&15 holds K k0..k0+7 & k0+16..k0+23 (lanes>=16: +8).
__device__ __forceinline__ v16bf load_bfrag(const ushort_t* __restrict__ w,
                                            int rowchunks, int row, int ks, int half) {
  const ushort_t* rowp = w + row * rowchunks * 8;
  int c0 = ks * 4 + half, sw = row & 15;
  union { v16bf v; u32x4 q[2]; } u;
  u.q[0] = *(const u32x4*)(rowp + (((c0    ) ^ sw) << 3));
  u.q[1] = *(const u32x4*)(rowp + (((c0 + 2) ^ sw) << 3));
  return u.v;
}

// A fragment (16x32 bf16) from a bf16 global row-major [M,ldk] matrix.
// ISA layout: lanes 0-15 hold K k0..k0+7 & k0+16..k0+23; lanes 16-31: +8.
__device__ __forceinline__ v16bf load_afrag_bf(const ushort_t* __restrict__ base,
                                               int ldk, int mrow, int k0, int half) {
  const ushort_t* p = base + mrow * ldk + k0 + half * 8;
  union { v16bf v; u32x4 q[2]; } u;
  u.q[0] = *(const u32x4*)p;
  u.q[1] = *(const u32x4*)(p + 16);
  return u.v;
}

// A fragment from an fp32 matrix (the input sequence), converted on the fly.
__device__ __forceinline__ v16bf load_afrag_f32(const float* __restrict__ base,
                                                int ldk, int mrow, int k0, int half) {
  const float* p = base + mrow * ldk + k0 + half * 8;
  v16bf v;
#pragma unroll
  for (int j = 0; j < 8; ++j) v[j] = (__bf16)p[j];
#pragma unroll
  for (int j = 0; j < 8; ++j) v[8 + j] = (__bf16)p[16 + j];
  return v;
}

// Grid-wide generation barrier (release/acquire at device scope).
__device__ __forceinline__ void grid_barrier(unsigned* bar) {
  __syncthreads();
  if (threadIdx.x == 0) {
    unsigned gen = __hip_atomic_load(&bar[1], __ATOMIC_RELAXED, __HIP_MEMORY_SCOPE_AGENT);
    unsigned prev = __hip_atomic_fetch_add(&bar[0], 1u, __ATOMIC_ACQ_REL,
                                           __HIP_MEMORY_SCOPE_AGENT);
    if (prev == NWG - 1) {
      __hip_atomic_store(&bar[0], 0u, __ATOMIC_RELAXED, __HIP_MEMORY_SCOPE_AGENT);
      __hip_atomic_fetch_add(&bar[1], 1u, __ATOMIC_RELEASE, __HIP_MEMORY_SCOPE_AGENT);
    } else {
      while (__hip_atomic_load(&bar[1], __ATOMIC_ACQUIRE,
                               __HIP_MEMORY_SCOPE_AGENT) == gen)
        __builtin_amdgcn_s_sleep(1);
    }
  }
  __syncthreads();
}

__global__ void slstm_prep(const float* __restrict__ bih0, const float* __restrict__ bhh0,
                           const float* __restrict__ bih1, const float* __restrict__ bhh1,
                           float* __restrict__ bs0, float* __restrict__ bs1,
                           ushort_t* __restrict__ h0bf, ushort_t* __restrict__ h1bf,
                           unsigned* __restrict__ bar) {
  int i = blockIdx.x * blockDim.x + threadIdx.x;
  int stride = gridDim.x * blockDim.x;
  for (int j = i; j < 4 * HDIM; j += stride) { bs0[j] = bih0[j] + bhh0[j];
                                               bs1[j] = bih1[j] + bhh1[j]; }
  for (int j = i; j < 2 * BATCH * HDIM; j += stride) { h0bf[j] = 0; h1bf[j] = 0; }
  if (i < 2) bar[i] = 0;
}

__global__ void __launch_bounds__(NTHR)
slstm_persistent(const float* __restrict__ xseq,
                 const float* __restrict__ Wih0, const float* __restrict__ Whh0,
                 const float* __restrict__ Wf0,  const float* __restrict__ bf0,
                 const float* __restrict__ Wih1, const float* __restrict__ Whh1,
                 const float* __restrict__ Wf1,  const float* __restrict__ bf1,
                 const float* __restrict__ bs0,  const float* __restrict__ bs1,
                 ushort_t* __restrict__ h0bf, ushort_t* __restrict__ h1bf,
                 unsigned* __restrict__ bar, float* __restrict__ out) {
  extern __shared__ __align__(16) char smem[];
  ushort_t* sW0 = (ushort_t*)smem;          // [64 rows][96 ch]  [Wih0|Whh0] slice
  ushort_t* sW1 = sW0 + SW0_US;             // [64 rows][128 ch] [Wih1|Whh1] slice
  ushort_t* sF0 = sW1 + SW1_US;             // [16 rows][64 ch]  Wf0 slice
  ushort_t* sF1 = sF0 + SWF_US;             // [16 rows][64 ch]  Wf1 slice
  float* gb = (float*)(sF1 + SWF_US);       // [4][64][17] gate staging (padded)
  float* fb = gb + 4 * 64 * 17;             // [64][17]    exp-gate staging

  const int tid = threadIdx.x;
  const int wg = blockIdx.x;
  const int colbase = wg * HS;

  // ---- one-time: load + convert + swizzle weight slices into LDS ----------
  for (int idx = tid; idx < 64 * W0_CH; idx += NTHR) {
    int row = idx / W0_CH, c = idx % W0_CH;
    int grow = (row >> 4) * HDIM + colbase + (row & 15);   // row in [4H] space
    const float* src = (c < 32) ? (Wih0 + grow * IDIM + c * 8)
                                : (Whh0 + grow * HDIM + (c - 32) * 8);
    store_wchunk(sW0 + row * W0_CH * 8, c, row, src);
  }
  for (int idx = tid; idx < 64 * W1_CH; idx += NTHR) {
    int row = idx / W1_CH, c = idx % W1_CH;
    int grow = (row >> 4) * HDIM + colbase + (row & 15);
    const float* src = (c < 64) ? (Wih1 + grow * HDIM + c * 8)
                                : (Whh1 + grow * HDIM + (c - 64) * 8);
    store_wchunk(sW1 + row * W1_CH * 8, c, row, src);
  }
  for (int idx = tid; idx < 16 * WF_CH; idx += NTHR) {
    int row = idx / WF_CH, c = idx % WF_CH;
    store_wchunk(sF0 + row * WF_CH * 8, c, row, Wf0 + (colbase + row) * HDIM + c * 8);
    store_wchunk(sF1 + row * WF_CH * 8, c, row, Wf1 + (colbase + row) * HDIM + c * 8);
  }
  __syncthreads();

  const int wave = tid >> 5, lane = tid & 31;
  const int half = lane >> 4, lrow = lane & 15;
  const int g = wave >> 1;                        // gate (i,f,g,o) for this wave
  const int mt0 = (wave & 1) * 2, mt1 = mt0 + 1;  // two batch tiles

  float c0r[4] = {0, 0, 0, 0}, c1r[4] = {0, 0, 0, 0};
  float h0r[4] = {0, 0, 0, 0}, h1r[4] = {0, 0, 0, 0};

#pragma unroll 1
  for (int t = 0; t < T_STEPS; ++t) {
    const int wp = t & 1, rp = wp ^ 1;
    const float* xt = xseq + (size_t)t * BATCH * IDIM;
    const ushort_t* h0rd = h0bf + rp * (BATCH * HDIM);
    const ushort_t* h1rd = h1bf + rp * (BATCH * HDIM);
    ushort_t* h0wr = h0bf + wp * (BATCH * HDIM);
    ushort_t* h1wr = h1bf + wp * (BATCH * HDIM);

    // Prefetch next step's x tile (speculative; dropped if OOB per ISA 10.5).
    if (t + 1 < T_STEPS) {
      const float* xn = xt + BATCH * IDIM;
      __builtin_prefetch(xn + (tid << 3), 0, 1);
      __builtin_prefetch(xn + (BATCH * IDIM / 2) + (tid << 3), 0, 1);
    }

    // ---- layer-0 gate GEMM: [x_t | h0(t-1)] @ [Wih0|Whh0]^T slice ---------
    {
      v8f a0 = {0, 0, 0, 0, 0, 0, 0, 0}, a1 = a0;
#pragma unroll
      for (int ks = 0; ks < 8; ++ks) {          // x part (fp32 -> bf16 cvt)
        v16bf b  = load_bfrag(sW0, W0_CH, g * 16 + lrow, ks, half);
        v16bf aA = load_afrag_f32(xt, IDIM, mt0 * 16 + lrow, ks * 32, half);
        v16bf aB = load_afrag_f32(xt, IDIM, mt1 * 16 + lrow, ks * 32, half);
        a0 = __builtin_amdgcn_wmma_f32_16x16x32_bf16(false, aA, false, b, (short)0, a0, false, false);
        a1 = __builtin_amdgcn_wmma_f32_16x16x32_bf16(false, aB, false, b, (short)0, a1, false, false);
      }
#pragma unroll 4
      for (int ks = 0; ks < 16; ++ks) {         // h0(t-1) part (bf16)
        v16bf b  = load_bfrag(sW0, W0_CH, g * 16 + lrow, 8 + ks, half);
        v16bf aA = load_afrag_bf(h0rd, HDIM, mt0 * 16 + lrow, ks * 32, half);
        v16bf aB = load_afrag_bf(h0rd, HDIM, mt1 * 16 + lrow, ks * 32, half);
        a0 = __builtin_amdgcn_wmma_f32_16x16x32_bf16(false, aA, false, b, (short)0, a0, false, false);
        a1 = __builtin_amdgcn_wmma_f32_16x16x32_bf16(false, aB, false, b, (short)0, a1, false, false);
      }
#pragma unroll
      for (int v = 0; v < 8; ++v) {
        gb[(g * 64 + mt0 * 16 + v + 8 * half) * 17 + lrow] = a0[v];
        gb[(g * 64 + mt1 * 16 + v + 8 * half) * 17 + lrow] = a1[v];
      }
    }
    __syncthreads();

    // ---- layer-0 pointwise: gates -> h0(t), c0'; publish bf16 h0 ----------
#pragma unroll
    for (int i = 0; i < 4; ++i) {
      int e = tid + i * NTHR, m = e >> 4, n = e & 15, gc = colbase + n;
      float gi = gb[(0 * 64 + m) * 17 + n] + bs0[0 * HDIM + gc];
      float gf = gb[(1 * 64 + m) * 17 + n] + bs0[1 * HDIM + gc];
      float gg = gb[(2 * 64 + m) * 17 + n] + bs0[2 * HDIM + gc];
      float go = gb[(3 * 64 + m) * 17 + n] + bs0[3 * HDIM + gc];
      float cv = sigf(gf) * c0r[i] + sigf(gi) * tanhf(gg);
      float hv = sigf(go) * tanhf(cv);
      c0r[i] = cv; h0r[i] = hv;
      h0wr[m * HDIM + gc] = f2bf(hv);
    }
    __threadfence();
    grid_barrier(bar);   // h0(t) visible everywhere

    // ---- layer-1 gate GEMM + Wf0 exp-gate GEMM ----------------------------
    {
      v8f a0 = {0, 0, 0, 0, 0, 0, 0, 0}, a1 = a0;
#pragma unroll 4
      for (int ks = 0; ks < 16; ++ks) {         // h0(t) part
        v16bf b  = load_bfrag(sW1, W1_CH, g * 16 + lrow, ks, half);
        v16bf aA = load_afrag_bf(h0wr, HDIM, mt0 * 16 + lrow, ks * 32, half);
        v16bf aB = load_afrag_bf(h0wr, HDIM, mt1 * 16 + lrow, ks * 32, half);
        a0 = __builtin_amdgcn_wmma_f32_16x16x32_bf16(false, aA, false, b, (short)0, a0, false, false);
        a1 = __builtin_amdgcn_wmma_f32_16x16x32_bf16(false, aB, false, b, (short)0, a1, false, false);
      }
#pragma unroll 4
      for (int ks = 0; ks < 16; ++ks) {         // h1(t-1) part
        v16bf b  = load_bfrag(sW1, W1_CH, g * 16 + lrow, 16 + ks, half);
        v16bf aA = load_afrag_bf(h1rd, HDIM, mt0 * 16 + lrow, ks * 32, half);
        v16bf aB = load_afrag_bf(h1rd, HDIM, mt1 * 16 + lrow, ks * 32, half);
        a0 = __builtin_amdgcn_wmma_f32_16x16x32_bf16(false, aA, false, b, (short)0, a0, false, false);
        a1 = __builtin_amdgcn_wmma_f32_16x16x32_bf16(false, aB, false, b, (short)0, a1, false, false);
      }
#pragma unroll
      for (int v = 0; v < 8; ++v) {
        gb[(g * 64 + mt0 * 16 + v + 8 * half) * 17 + lrow] = a0[v];
        gb[(g * 64 + mt1 * 16 + v + 8 * half) * 17 + lrow] = a1[v];
      }
    }
    if (wave < 4) {                       // f0 = h0(t) @ Wf0^T slice
      v8f af = {0, 0, 0, 0, 0, 0, 0, 0};
#pragma unroll 4
      for (int ks = 0; ks < 16; ++ks) {
        v16bf b = load_bfrag(sF0, WF_CH, lrow, ks, half);
        v16bf a = load_afrag_bf(h0wr, HDIM, wave * 16 + lrow, ks * 32, half);
        af = __builtin_amdgcn_wmma_f32_16x16x32_bf16(false, a, false, b, (short)0, af, false, false);
      }
#pragma unroll
      for (int v = 0; v < 8; ++v)
        fb[(wave * 16 + v + 8 * half) * 17 + lrow] = af[v];
    }
    __syncthreads();

    // ---- layer-1 pointwise + layer-0 exp gate; publish h1, write out_seq --
#pragma unroll
    for (int i = 0; i < 4; ++i) {
      int e = tid + i * NTHR, m = e >> 4, n = e & 15, gc = colbase + n;
      float gi = gb[(0 * 64 + m) * 17 + n] + bs1[0 * HDIM + gc];
      float gf = gb[(1 * 64 + m) * 17 + n] + bs1[1 * HDIM + gc];
      float gg = gb[(2 * 64 + m) * 17 + n] + bs1[2 * HDIM + gc];
      float go = gb[(3 * 64 + m) * 17 + n] + bs1[3 * HDIM + gc];
      float cv = sigf(gf) * c1r[i] + sigf(gi) * tanhf(gg);
      float hv = sigf(go) * tanhf(cv);
      c1r[i] = cv; h1r[i] = hv;
      h1wr[m * HDIM + gc] = f2bf(hv);
      out[(size_t)m * (T_STEPS * HDIM) + (size_t)t * HDIM + gc] = hv;
      // sLSTM exp forget gate, layer 0 (Wi term is *0 in reference: dropped)
      c0r[i] = __expf(fb[m * 17 + n] + bf0[gc]) * c0r[i];
    }
    __threadfence();
    grid_barrier(bar);   // h1(t) visible everywhere

    // ---- layer-1 exp gate: c1 <- exp(h1 @ Wf1^T + bf1) * c1 ---------------
    if (wave < 4) {
      v8f af = {0, 0, 0, 0, 0, 0, 0, 0};
#pragma unroll 4
      for (int ks = 0; ks < 16; ++ks) {
        v16bf b = load_bfrag(sF1, WF_CH, lrow, ks, half);
        v16bf a = load_afrag_bf(h1wr, HDIM, wave * 16 + lrow, ks * 32, half);
        af = __builtin_amdgcn_wmma_f32_16x16x32_bf16(false, a, false, b, (short)0, af, false, false);
      }
#pragma unroll
      for (int v = 0; v < 8; ++v)
        fb[(wave * 16 + v + 8 * half) * 17 + lrow] = af[v];
    }
    __syncthreads();
#pragma unroll
    for (int i = 0; i < 4; ++i) {
      int e = tid + i * NTHR, m = e >> 4, n = e & 15, gc = colbase + n;
      c1r[i] = __expf(fb[m * 17 + n] + bf1[gc]) * c1r[i];
    }
  }

  // ---- final hidden: (2B, 2H) = rows [h0|c0] then [h1|c1] -----------------
  const size_t hid = (size_t)BATCH * T_STEPS * HDIM;
#pragma unroll
  for (int i = 0; i < 4; ++i) {
    int e = tid + i * NTHR, m = e >> 4, n = e & 15, gc = colbase + n;
    out[hid + (size_t)m * (2 * HDIM) + gc] = h0r[i];
    out[hid + (size_t)m * (2 * HDIM) + HDIM + gc] = c0r[i];
    out[hid + (size_t)(BATCH + m) * (2 * HDIM) + gc] = h1r[i];
    out[hid + (size_t)(BATCH + m) * (2 * HDIM) + HDIM + gc] = c1r[i];
  }
}

extern "C" void kernel_launch(void* const* d_in, const int* in_sizes, int n_in,
                              void* d_out, int out_size, void* d_ws, size_t ws_size,
                              hipStream_t stream) {
  (void)in_sizes; (void)n_in; (void)out_size; (void)ws_size;
  const float* xseq = (const float*)d_in[0];
  const float* Wih0 = (const float*)d_in[1];
  const float* Whh0 = (const float*)d_in[2];
  const float* bih0 = (const float*)d_in[3];
  const float* bhh0 = (const float*)d_in[4];
  const float* Wf0  = (const float*)d_in[5];
  const float* bf0  = (const float*)d_in[6];
  const float* Wih1 = (const float*)d_in[9];
  const float* Whh1 = (const float*)d_in[10];
  const float* bih1 = (const float*)d_in[11];
  const float* bhh1 = (const float*)d_in[12];
  const float* Wf1  = (const float*)d_in[13];
  const float* bf1  = (const float*)d_in[14];

  // workspace: bias sums, double-buffered bf16 h arrays, barrier
  float* bs0 = (float*)d_ws;
  float* bs1 = bs0 + 4 * HDIM;
  ushort_t* h0bf = (ushort_t*)(bs1 + 4 * HDIM);
  ushort_t* h1bf = h0bf + 2 * BATCH * HDIM;
  unsigned* bar = (unsigned*)(h1bf + 2 * BATCH * HDIM);

  hipLaunchKernelGGL(slstm_prep, dim3(64), dim3(256), 0, stream,
                     bih0, bhh0, bih1, bhh1, bs0, bs1, h0bf, h1bf, bar);

  hipLaunchKernelGGL(slstm_persistent, dim3(NWG), dim3(NTHR), LDS_BYTES, stream,
                     xseq, Wih0, Whh0, Wf0, bf0, Wih1, Whh1, Wf1, bf1,
                     bs0, bs1, h0bf, h1bf, bar, (float*)d_out);
}
